// FGAM_4217657884895
// MI455X (gfx1250) — matrix operand order repaired and checked by reading, not compiled, not run on hardware
//
#include <hip/hip_runtime.h>

#define DEVINL __device__ __forceinline__

constexpr int B_   = 4096;
constexpr int F_   = 64;
constexpr int E_   = 256;
constexpr int H_   = 128;
constexpr int S_   = 128;
constexpr int NHID = 3;

typedef __attribute__((ext_vector_type(16))) __bf16 v16bf;
typedef __attribute__((ext_vector_type(8)))  float  v8f;

struct Pack32 { uint4 lo; uint4 hi; };   // 32 bytes == v16bf

DEVINL unsigned short f2bf(float x) {
    unsigned u = __builtin_bit_cast(unsigned, x);
    unsigned r = u + 0x7FFFu + ((u >> 16) & 1u);   // round-to-nearest-even
    return (unsigned short)(r >> 16);
}
DEVINL unsigned pk2(float a, float b) {
    return (unsigned)f2bf(a) | ((unsigned)f2bf(b) << 16);
}
DEVINL float bf2f(unsigned short u) {
    return __builtin_bit_cast(float, ((unsigned)u) << 16);
}

// ---------------- weight pre-conversion: f32 -> bf16 (one pass) -------------
__global__ void __launch_bounds__(256)
cvt_bf16_kernel(const float* __restrict__ src, unsigned short* __restrict__ dst,
                int n)
{
    int i = (blockIdx.x * 256 + threadIdx.x) * 4;
    if (i + 3 < n) {
        float4 v = *(const float4*)(src + i);
        unsigned* p = (unsigned*)(dst + i);
        p[0] = pk2(v.x, v.y);
        p[1] = pk2(v.z, v.w);
    } else {
        for (int k = i; k < n; ++k) dst[k] = f2bf(src[k]);
    }
}

// GEMM inner loop: A (M-tile x K, bf16, LDS row-major) x B^T, where the weight
// matrix Wg is row-major (n, k) bf16 in global (pre-converted, L2-resident).
// Each wave owns one 16-row M tile and NT 16-col N tiles.
// A frag (16x32 bf16): lane l15=M row; K runs [kb+8*hi .. +7], [kb+16+8*hi .. +7].
// B frag (32x16 bf16): lane l15=N col; K runs [kb+8*hi .. +7], [kb+16+8*hi .. +7].
template <int NT>
DEVINL void gemm_bf16(const unsigned short* __restrict__ aLds, int aStride, int K,
                      const unsigned short* __restrict__ Wg, int ldW,
                      int m0, int nBase, int lane, v8f* acc)
{
    const int l15 = lane & 15;
    const int hi  = lane >> 4;
#pragma unroll 2
    for (int kb = 0; kb < K; kb += 32) {
        const unsigned short* ap = aLds + (m0 + l15) * aStride + kb + 8 * hi;
        Pack32 pa;
        pa.lo = *(const uint4*)(ap);
        pa.hi = *(const uint4*)(ap + 16);
        v16bf aF = __builtin_bit_cast(v16bf, pa);
#pragma unroll
        for (int nt = 0; nt < NT; ++nt) {
            const unsigned short* bp =
                Wg + (size_t)(nBase + nt * 16 + l15) * ldW + kb + 8 * hi;
            Pack32 pb;
            pb.lo = *(const uint4*)(bp);
            pb.hi = *(const uint4*)(bp + 16);
            v16bf bF = __builtin_bit_cast(v16bf, pb);
            acc[nt] = __builtin_amdgcn_wmma_f32_16x16x32_bf16(
                false, aF, false, bF, (short)0, acc[nt], false, false);
        }
    }
}

// ---------------- Kernel 1: per-feature MLP ensemble -> x (B,F) -------------
// grid (B/64, F), 256 threads (8 waves). Wave w: M tile = w>>1, N half = w&1.
__global__ void __launch_bounds__(256)
gam_feature_kernel(const float* __restrict__ tv,
                   const float* __restrict__ W0, const float* __restrict__ b0,
                   const unsigned short* __restrict__ Whb,
                   const float* __restrict__ bh,
                   const unsigned short* __restrict__ Wpb,
                   const float* __restrict__ bp,
                   const float* __restrict__ Wo, const float* __restrict__ bo,
                   float* __restrict__ x_ws)
{
    extern __shared__ unsigned short smem[];
    constexpr int AST = E_ + 8;                 // 264 ushorts/row (16B-aligned)
    unsigned short* bufA = smem;
    unsigned short* bufB = smem + 64 * AST;
    float* xacc = (float*)(smem + 2 * 64 * AST);

    const int f     = blockIdx.y;
    const int bBase = blockIdx.x * 64;
    const int tid   = threadIdx.x;
    const int lane  = tid & 31;
    const int wave  = tid >> 5;
    const int m0    = (wave >> 1) * 16;
    const int nHalf = wave & 1;
    const int hi    = lane >> 4;

    if (tid < 64) xacc[tid] = 0.0f;

    // ---- h0 = relu(tv[b,f] * W0[f,:] + b0[f,:]) -> bufA (bf16) ----
    {
        int row = tid >> 2, q = tid & 3;
        float tvv = tv[(size_t)(bBase + row) * F_ + f];
        const float4* w4 = (const float4*)(W0 + (size_t)f * E_);
        const float4* c4 = (const float4*)(b0 + (size_t)f * E_);
        unsigned short* dst = bufA + row * AST;
#pragma unroll
        for (int j = 0; j < 16; ++j) {
            int c = q * 64 + j * 4;
            float4 w = w4[c >> 2], bb = c4[c >> 2];
            float v0 = fmaxf(tvv * w.x + bb.x, 0.0f);
            float v1 = fmaxf(tvv * w.y + bb.y, 0.0f);
            float v2 = fmaxf(tvv * w.z + bb.z, 0.0f);
            float v3 = fmaxf(tvv * w.w + bb.w, 0.0f);
            unsigned* p = (unsigned*)(dst + c);
            p[0] = pk2(v0, v1);
            p[1] = pk2(v2, v3);
        }
    }
    __syncthreads();

    unsigned short* aIn  = bufA;
    unsigned short* aOut = bufB;

    // ---- 3 hidden layers: h = relu(h @ Wh[l,f]^T + bh[l,f]) ----
    for (int l = 0; l < NHID; ++l) {
        const unsigned short* Wl = Whb + (((size_t)l * F_ + f) * E_) * E_;
        const float* bl = bh + ((size_t)l * F_ + f) * E_;
        v8f acc[8] = {};
        gemm_bf16<8>(aIn, AST, E_, Wl, E_, m0, nHalf * 128, lane, acc);
#pragma unroll
        for (int nt = 0; nt < 8; ++nt) {
            int n = nHalf * 128 + nt * 16 + (lane & 15);
            float bias = bl[n];
#pragma unroll
            for (int r = 0; r < 8; ++r) {
                float v = fmaxf(acc[nt][r] + bias, 0.0f);
                aOut[(m0 + r + 8 * hi) * AST + n] = f2bf(v);
            }
        }
        __syncthreads();
        unsigned short* t = aIn; aIn = aOut; aOut = t;
    }

    // ---- p = relu(h @ Wp[f]^T + bp[f]); x = p . Wo[f] + bo[f] ----
    {
        const unsigned short* Wlp = Wpb + (size_t)f * H_ * E_;
        const float* blp = bp + (size_t)f * H_;
        const float* wo  = Wo + (size_t)f * H_;
        v8f acc[4] = {};
        gemm_bf16<4>(aIn, AST, E_, Wlp, E_, m0, nHalf * 64, lane, acc);
        float part[8] = {0, 0, 0, 0, 0, 0, 0, 0};
#pragma unroll
        for (int nt = 0; nt < 4; ++nt) {
            int n = nHalf * 64 + nt * 16 + (lane & 15);
            float bias = blp[n];
            float won  = wo[n];
#pragma unroll
            for (int r = 0; r < 8; ++r) {
                float v = fmaxf(acc[nt][r] + bias, 0.0f);
                part[r] += v * won;
            }
        }
#pragma unroll
        for (int r = 0; r < 8; ++r) {
            float v = part[r];
            v += __shfl_xor(v, 1);
            v += __shfl_xor(v, 2);
            v += __shfl_xor(v, 4);
            v += __shfl_xor(v, 8);
            if ((lane & 15) == 0)
                atomicAdd(&xacc[m0 + r + 8 * hi], v);   // ds_add_f32
        }
    }
    __syncthreads();
    if (tid < 64)
        x_ws[(size_t)(bBase + tid) * F_ + f] = xacc[tid] + bo[f];
}

// ---------------- Kernel 2: weighting head -> logits column -----------------
// grid (B/64), 256 threads. One launch per head.
__global__ void __launch_bounds__(256)
gam_head_kernel(const float* __restrict__ stat, const float* __restrict__ x_ws,
                const unsigned short* __restrict__ Wx0, const float* __restrict__ bx0,
                const unsigned short* __restrict__ Wx1, const float* __restrict__ bx1,
                const unsigned short* __restrict__ Ww,  const float* __restrict__ bw,
                const float* __restrict__ Wb,  const float* __restrict__ bb,
                float* __restrict__ out, int head)
{
    constexpr int WST = H_ + 8;                 // 136 ushorts/row
    __shared__ unsigned short sA[64 * WST];
    __shared__ unsigned short sW0[64 * WST];
    __shared__ unsigned short sW1[64 * WST];
    __shared__ float logacc[64];

    const int bBase = blockIdx.x * 64;
    const int tid   = threadIdx.x;
    const int lane  = tid & 31;
    const int wave  = tid >> 5;
    const int m0    = (wave >> 1) * 16;
    const int nHalf = wave & 1;
    const int hi    = lane >> 4;

    if (tid < 64) logacc[tid] = 0.0f;

    // stage static tile (64 x S) as bf16
    {
        int row = tid >> 2, q = tid & 3;
        const float4* s4 = (const float4*)(stat + (size_t)(bBase + row) * S_);
        unsigned short* dst = sA + row * WST;
#pragma unroll
        for (int j = 0; j < 8; ++j) {
            int c = q * 32 + j * 4;
            float4 v = s4[c >> 2];
            unsigned* p = (unsigned*)(dst + c);
            p[0] = pk2(v.x, v.y);
            p[1] = pk2(v.z, v.w);
        }
    }
    __syncthreads();

    // w0 = relu(static @ Wx0^T + bx0)
    {
        v8f acc[4] = {};
        gemm_bf16<4>(sA, WST, S_, Wx0, S_, m0, nHalf * 64, lane, acc);
#pragma unroll
        for (int nt = 0; nt < 4; ++nt) {
            int n = nHalf * 64 + nt * 16 + (lane & 15);
            float bias = bx0[n];
#pragma unroll
            for (int r = 0; r < 8; ++r) {
                float v = fmaxf(acc[nt][r] + bias, 0.0f);
                sW0[(m0 + r + 8 * hi) * WST + n] = f2bf(v);
            }
        }
    }
    __syncthreads();

    // w1 = relu(w0 @ Wx1^T + bx1)
    {
        v8f acc[4] = {};
        gemm_bf16<4>(sW0, WST, H_, Wx1, H_, m0, nHalf * 64, lane, acc);
#pragma unroll
        for (int nt = 0; nt < 4; ++nt) {
            int n = nHalf * 64 + nt * 16 + (lane & 15);
            float bias = bx1[n];
#pragma unroll
            for (int r = 0; r < 8; ++r) {
                float v = fmaxf(acc[nt][r] + bias, 0.0f);
                sW1[(m0 + r + 8 * hi) * WST + n] = f2bf(v);
            }
        }
    }
    __syncthreads();

    // weight = w1 @ Ww^T + bw (N = F = 64), fused with sum_f x * weight
    {
        v8f acc[2] = {};
        gemm_bf16<2>(sW1, WST, H_, Ww, H_, m0, nHalf * 32, lane, acc);
        float part[8] = {0, 0, 0, 0, 0, 0, 0, 0};
#pragma unroll
        for (int nt = 0; nt < 2; ++nt) {
            int n = nHalf * 32 + nt * 16 + (lane & 15);   // feature index
            float bwn = bw[n];
#pragma unroll
            for (int r = 0; r < 8; ++r) {
                int m = m0 + r + 8 * hi;
                float wv = acc[nt][r] + bwn;
                float xv = x_ws[(size_t)(bBase + m) * F_ + n];
                part[r] += wv * xv;
            }
        }
#pragma unroll
        for (int r = 0; r < 8; ++r) {
            float v = part[r];
            v += __shfl_xor(v, 1);
            v += __shfl_xor(v, 2);
            v += __shfl_xor(v, 4);
            v += __shfl_xor(v, 8);
            if ((lane & 15) == 0)
                atomicAdd(&logacc[m0 + r + 8 * hi], v);
        }
    }

    // bias = w1 @ Wb^T + bb  (N = 1): dot per row from LDS
    {
        int row = tid >> 2, q = tid & 3;
        const unsigned short* src = sW1 + row * WST;
        float s = 0.0f;
#pragma unroll
        for (int k = 0; k < 32; ++k) {
            int kk = q * 32 + k;
            s += bf2f(src[kk]) * Wb[kk];
        }
        if (q == 0) s += bb[0];
        atomicAdd(&logacc[row], s);
    }
    __syncthreads();
    if (tid < 64)
        out[(size_t)(bBase + tid) * 2 + head] = logacc[tid];
}

extern "C" void kernel_launch(void* const* d_in, const int* in_sizes, int n_in,
                              void* d_out, int out_size, void* d_ws, size_t ws_size,
                              hipStream_t stream) {
    const float* stat = (const float*)d_in[0];
    const float* tv   = (const float*)d_in[1];
    const float* W0   = (const float*)d_in[2];
    const float* b0   = (const float*)d_in[3];
    const float* Wh   = (const float*)d_in[4];
    const float* bh   = (const float*)d_in[5];
    const float* Wp   = (const float*)d_in[6];
    const float* bp   = (const float*)d_in[7];
    const float* Wo   = (const float*)d_in[8];
    const float* bo   = (const float*)d_in[9];
    const float* Wa0  = (const float*)d_in[10];
    const float* ba0  = (const float*)d_in[11];
    const float* Wa1  = (const float*)d_in[12];
    const float* ba1  = (const float*)d_in[13];
    const float* Wwt  = (const float*)d_in[14];
    const float* bwt  = (const float*)d_in[15];
    const float* Wbb  = (const float*)d_in[16];
    const float* bbb  = (const float*)d_in[17];
    const float* Wc0  = (const float*)d_in[18];
    const float* bc0  = (const float*)d_in[19];
    const float* Wc1  = (const float*)d_in[20];
    const float* bc1  = (const float*)d_in[21];
    const float* Wwt2 = (const float*)d_in[22];
    const float* bwt2 = (const float*)d_in[23];
    const float* Wbb2 = (const float*)d_in[24];
    const float* bbb2 = (const float*)d_in[25];

    float* out = (float*)d_out;

    // ---- workspace layout ----
    char* ws = (char*)d_ws;
    float* x_ws = (float*)ws;                                 // B*F f32 (1 MB)
    size_t off = (size_t)B_ * F_ * sizeof(float);
    unsigned short* Whb = (unsigned short*)(ws + off);        // 24 MB bf16
    off += (size_t)NHID * F_ * E_ * E_ * sizeof(unsigned short);
    unsigned short* Wpb = (unsigned short*)(ws + off);        // 4 MB bf16
    off += (size_t)F_ * H_ * E_ * sizeof(unsigned short);
    unsigned short* Wa0b = (unsigned short*)(ws + off); off += (size_t)H_ * S_ * 2;
    unsigned short* Wa1b = (unsigned short*)(ws + off); off += (size_t)H_ * H_ * 2;
    unsigned short* Wwtb = (unsigned short*)(ws + off); off += (size_t)F_ * H_ * 2;
    unsigned short* Wc0b = (unsigned short*)(ws + off); off += (size_t)H_ * S_ * 2;
    unsigned short* Wc1b = (unsigned short*)(ws + off); off += (size_t)H_ * H_ * 2;
    unsigned short* Wwt2b = (unsigned short*)(ws + off);

    // ---- pre-convert all GEMM weights to bf16 (amortized over 64 tiles) ----
    auto cvt = [&](const float* s, unsigned short* d, int n) {
        int blocks = (n / 4 + 255) / 256;
        cvt_bf16_kernel<<<blocks, 256, 0, stream>>>(s, d, n);
    };
    cvt(Wh,   Whb,   NHID * F_ * E_ * E_);
    cvt(Wp,   Wpb,   F_ * H_ * E_);
    cvt(Wa0,  Wa0b,  H_ * S_);
    cvt(Wa1,  Wa1b,  H_ * H_);
    cvt(Wwt,  Wwtb,  F_ * H_);
    cvt(Wc0,  Wc0b,  H_ * S_);
    cvt(Wc1,  Wc1b,  H_ * H_);
    cvt(Wwt2, Wwt2b, F_ * H_);

    size_t lds1 = (size_t)2 * 64 * (E_ + 8) * sizeof(unsigned short)
                + 64 * sizeof(float);
    dim3 g1(B_ / 64, F_);
    gam_feature_kernel<<<g1, 256, lds1, stream>>>(tv, W0, b0, Whb, bh, Wpb, bp,
                                                  Wo, bo, x_ws);
    gam_head_kernel<<<B_ / 64, 256, 0, stream>>>(stat, x_ws, Wa0b, ba0, Wa1b, ba1,
                                                 Wwtb, bwt, Wbb, bbb, out, 0);
    gam_head_kernel<<<B_ / 64, 256, 0, stream>>>(stat, x_ws, Wc0b, bc0, Wc1b, bc1,
                                                 Wwt2b, bwt2, Wbb2, bbb2, out, 1);
}
